// BiMambaBlock_14353780703878
// MI455X (gfx1250) — compile-verified, matrix operand order checked by reading, not run on hardware
//
#include <hip/hip_runtime.h>
#include <hip/hip_bf16.h>
#include <math.h>

// Problem constants (match reference)
constexpr int BATCH = 4;
constexpr int LSEQ  = 4096;
constexpr int C_IN  = 12;
constexpr int C_OUT = 4;
constexpr int HDIM  = 32;
constexpr int DIN   = 64;
constexpr int NST   = 16;
constexpr int KCONV = 4;
constexpr int NBLK  = 6;
constexpr int RROWS = BATCH * LSEQ;      // 16384
constexpr int NCH   = 16;                 // scan chunks per sequence
constexpr int CLEN  = LSEQ / NCH;         // 256

typedef _Float16 v16h __attribute__((ext_vector_type(16)));
typedef __fp16   v2fp __attribute__((ext_vector_type(2)));   // cvt_pkrtz result type
typedef float    v8f  __attribute__((ext_vector_type(8)));

__device__ __forceinline__ float sigm(float x) { return 1.0f / (1.0f + __expf(-x)); }

// ---------------------------------------------------------------------------
// Strided WMMA GEMM:  Y[b,l,o] = act( sum_k X[b, lr, k] * W[o,k] + bias[o] )
// lr = rev ? LSEQ-1-l : l.  One wave computes a 16(M) x 16(N) tile, K stepped by
// 32 via v_wmma_f32_16x16x32_f16; KDIM is compile-time so the k-loop fully
// unrolls and all global loads issue branch-free (clamped index + cndmask zero),
// letting the compiler batch loads under one s_wait.
// A layout (16-bit 16x32, ISA 7.12.2): lane m = lane&15; lane group g = lane>>4
//   vgpr pair j holds K = (j<4 ? 2j+8g : 16+2(j-4)+8g) .. +1
// C/D layout: vgpr i -> row i + 8g, col = lane&15.
// ---------------------------------------------------------------------------
template <int KDIM, bool KCONTIG>
__global__ void gemm_wmma(const float* __restrict__ X, long xB, long xL, long xK, int rev,
                          const float* __restrict__ W, const float* __restrict__ bias,
                          float* __restrict__ Y, long yB, long yL, long yO,
                          int O, int act)
{
  const int lane = threadIdx.x & 31;
  const int wave = threadIdx.x >> 5;
  const int rt   = blockIdx.x * 2 + wave;   // row tile (16 rows)
  const int ot   = blockIdx.y;              // col tile (16 cols)
  const int mn   = lane & 15;
  const int g    = lane >> 4;

  const int r  = rt * 16 + mn;              // A-row this lane feeds
  const int ab = r / LSEQ;
  const int al = r % LSEQ;
  const int lr = rev ? (LSEQ - 1 - al) : al;
  const long abase = (long)ab * xB + (long)lr * xL;

  const int  o      = ot * 16 + mn;         // B-col this lane feeds
  const bool ovalid = (o < O);
  const float* Wrow = W + (long)(ovalid ? o : 0) * KDIM;

  v8f acc = {};
#pragma unroll
  for (int kb = 0; kb < KDIM; kb += 32) {
    v16h a, bf;
#pragma unroll
    for (int j = 0; j < 8; ++j) {
      const int  k0 = kb + ((j < 4) ? (2 * j + 8 * g) : (16 + 2 * (j - 4) + 8 * g));
      // Pair-granular k validity (KDIM is even; pairs are fully valid/invalid).
      const bool kv = (KDIM % 32 == 0) || (k0 + 1 < KDIM);
      const int  kc = kv ? k0 : 0;          // clamped, always-legal index
      float a0, a1;
      if (KCONTIG) {
        const float2 av = *(const float2*)(X + abase + kc);   // 8B-aligned
        a0 = av.x; a1 = av.y;
      } else {
        a0 = X[abase + (long)kc * xK];
        a1 = X[abase + (long)(kc + 1) * xK];
      }
      const float2 wv = *(const float2*)(Wrow + kc);
      float b0 = wv.x, b1 = wv.y;
      if (!kv)     { a0 = 0.f; a1 = 0.f; b0 = 0.f; b1 = 0.f; }
      if (!ovalid) { b0 = 0.f; b1 = 0.f; }
      const v2fp pa = __builtin_amdgcn_cvt_pkrtz(a0, a1);
      const v2fp pb = __builtin_amdgcn_cvt_pkrtz(b0, b1);
      a [2 * j] = (_Float16)pa.x;  a [2 * j + 1] = (_Float16)pa.y;
      bf[2 * j] = (_Float16)pb.x;  bf[2 * j + 1] = (_Float16)pb.y;
    }
    acc = __builtin_amdgcn_wmma_f32_16x16x32_f16(false, a, false, bf, (short)0, acc,
                                                 false, false);
  }

  if (ovalid) {
    const float bv = bias ? bias[o] : 0.f;
#pragma unroll
    for (int i = 0; i < 8; ++i) {
      const int rr = rt * 16 + i + g * 8;
      const int wb = rr / LSEQ;
      const int wl = rr % LSEQ;
      float v = acc[i] + bv;
      if (act == 1) v = sigm(v);
      Y[(long)wb * yB + (long)wl * yL + (long)o * yO] = v;
    }
  }
}

// Depthwise causal conv (K=4) + SiLU.  xm = xz[:, :DIN] (row stride 128).
__global__ void conv_silu_kernel(const float* __restrict__ xz, const float* __restrict__ cw,
                                 const float* __restrict__ cb, float* __restrict__ xc)
{
  const int idx = blockIdx.x * blockDim.x + threadIdx.x;
  if (idx >= RROWS * DIN) return;
  const int d = idx & 63;
  const int r = idx >> 6;
  const int l = r % LSEQ;
  const int b = r / LSEQ;
  float s = cb[d];
#pragma unroll
  for (int k = 0; k < KCONV; ++k) {
    const int ls0 = l - (KCONV - 1) + k;
    const int ls  = (ls0 >= 0) ? ls0 : 0;            // clamp, mask arithmetically
    const float v = xz[((long)(b * LSEQ + ls)) * 128 + d];
    s += ((ls0 >= 0) ? v : 0.f) * cw[d * KCONV + k];
  }
  xc[(long)r * DIN + d] = s * sigm(s);
}

// dt = softplus(x_dbl[:, :2] @ dtw^T + dtb)
__global__ void dt_kernel(const float* __restrict__ xdbl, const float* __restrict__ dtw,
                          const float* __restrict__ dtbias, float* __restrict__ dtb)
{
  const int idx = blockIdx.x * blockDim.x + threadIdx.x;
  if (idx >= RROWS * DIN) return;
  const int d = idx & 63;
  const long r = idx >> 6;
  float v = xdbl[r * 34 + 0] * dtw[d * 2 + 0] + xdbl[r * 34 + 1] * dtw[d * 2 + 1] + dtbias[d];
  dtb[r * DIN + d] = (v > 20.f) ? v : log1pf(__expf(v));
}

// Chunked scan pass 1: per (b,chunk) run scan with zero init; emit final state
// and cumulative decay product.  64 threads = one d each, 16 n-states in regs.
__global__ void scan_pass1(const float* __restrict__ dtb, const float* __restrict__ xcb,
                           const float* __restrict__ xdbl, const float* __restrict__ Alog,
                           float* __restrict__ hfin, float* __restrict__ pA)
{
  __shared__ float sB[64 * 16];
  const int b = blockIdx.x / NCH;
  const int c = blockIdx.x % NCH;
  const int d = threadIdx.x;
  float Am[NST], h[NST], P[NST];
#pragma unroll
  for (int n = 0; n < NST; ++n) { Am[n] = -__expf(Alog[d * NST + n]); h[n] = 0.f; P[n] = 1.f; }
  const int rbase = b * LSEQ + c * CLEN;
  for (int t0 = 0; t0 < CLEN; t0 += 64) {
    __syncthreads();
#pragma unroll
    for (int i = 0; i < 16; ++i) {
      const int e = i * 64 + d;
      sB[e] = xdbl[(long)(rbase + t0 + (e >> 4)) * 34 + 2 + (e & 15)];
    }
    __syncthreads();
    for (int tt = 0; tt < 64; ++tt) {
      const long r   = rbase + t0 + tt;
      const float dt = dtb[r * DIN + d];
      const float dx = dt * xcb[r * DIN + d];
#pragma unroll
      for (int n = 0; n < NST; ++n) {
        const float aa = __expf(dt * Am[n]);
        h[n] = aa * h[n] + dx * sB[tt * 16 + n];
        P[n] *= aa;
      }
    }
  }
  const long o = (long)blockIdx.x * 1024 + d * NST;
#pragma unroll
  for (int n = 0; n < NST; ++n) { hfin[o + n] = h[n]; pA[o + n] = P[n]; }
}

// Pass 2: tiny serial chunk-prefix combine: s_{c+1} = P_c * s_c + hfin_c.
__global__ void scan_pass2(const float* __restrict__ hfin, const float* __restrict__ pA,
                           float* __restrict__ hst)
{
  const int idx = blockIdx.x * blockDim.x + threadIdx.x;  // BATCH*1024
  if (idx >= BATCH * 1024) return;
  const int b = idx >> 10, dn = idx & 1023;
  float s = 0.f;
  for (int c = 0; c < NCH; ++c) {
    const long o = (long)(b * NCH + c) * 1024 + dn;
    hst[o] = s;
    s = pA[o] * s + hfin[o];
  }
}

// Pass 3: replay each chunk from its true initial state; fuse +x*D and SiLU(z) gate.
__global__ void scan_pass3(const float* __restrict__ dtb, const float* __restrict__ xcb,
                           const float* __restrict__ xdbl, const float* __restrict__ xz,
                           const float* __restrict__ Alog, const float* __restrict__ Dp,
                           const float* __restrict__ hst, float* __restrict__ yb)
{
  __shared__ float sB[64 * 16];
  __shared__ float sC[64 * 16];
  const int b = blockIdx.x / NCH;
  const int c = blockIdx.x % NCH;
  const int d = threadIdx.x;
  float Am[NST], h[NST];
  const long o0 = (long)blockIdx.x * 1024 + d * NST;
#pragma unroll
  for (int n = 0; n < NST; ++n) { Am[n] = -__expf(Alog[d * NST + n]); h[n] = hst[o0 + n]; }
  const float Dd = Dp[d];
  const int rbase = b * LSEQ + c * CLEN;
  for (int t0 = 0; t0 < CLEN; t0 += 64) {
    __syncthreads();
#pragma unroll
    for (int i = 0; i < 16; ++i) {
      const int e = i * 64 + d;
      const long rr = (long)(rbase + t0 + (e >> 4)) * 34;
      sB[e] = xdbl[rr + 2 + (e & 15)];
      sC[e] = xdbl[rr + 18 + (e & 15)];
    }
    __syncthreads();
    for (int tt = 0; tt < 64; ++tt) {
      const long r   = rbase + t0 + tt;
      const float dt = dtb[r * DIN + d];
      const float sx = xcb[r * DIN + d];
      const float dx = dt * sx;
      float acc = 0.f;
#pragma unroll
      for (int n = 0; n < NST; ++n) {
        const float aa = __expf(dt * Am[n]);
        h[n] = aa * h[n] + dx * sB[tt * 16 + n];
        acc += h[n] * sC[tt * 16 + n];
      }
      float yv = acc + sx * Dd;
      const float zv = xz[r * 128 + DIN + d];
      yv *= zv * sigm(zv);
      yb[r * DIN + d] = yv;
    }
  }
}

// ---------------------------------------------------------------------------
extern "C" void kernel_launch(void* const* d_in, const int* in_sizes, int n_in,
                              void* d_out, int out_size, void* d_ws, size_t ws_size,
                              hipStream_t stream) {
  const float* x        = (const float*)d_in[0];   // (B, CIN, L)
  const float* proj1_w  = (const float*)d_in[1];   // (32, 12)
  const float* proj1_b  = (const float*)d_in[2];   // (32)
  const float* projr_w  = (const float*)d_in[3];   // (5, 32, 64)
  const float* projr_b  = (const float*)d_in[4];   // (5, 32)
  const float* in_proj  = (const float*)d_in[5];   // (12, 128, 32)
  const float* conv_w   = (const float*)d_in[6];   // (12, 64, 4)
  const float* conv_b   = (const float*)d_in[7];   // (12, 64)
  const float* xproj_w  = (const float*)d_in[8];   // (12, 34, 64)
  const float* dtproj_w = (const float*)d_in[9];   // (12, 64, 2)
  const float* dtproj_b = (const float*)d_in[10];  // (12, 64)
  const float* A_log    = (const float*)d_in[11];  // (12, 64, 16)
  const float* Dparam   = (const float*)d_in[12];  // (12, 64)
  const float* out_proj = (const float*)d_in[13];  // (12, 32, 64)
  const float* Wout     = (const float*)d_in[14];  // (4, 64)
  const float* bout     = (const float*)d_in[15];  // (4)
  float* out = (float*)d_out;                      // (B, L, 4)

  // Workspace carve-up (~35 MB of floats; harness scratch).
  float* ws   = (float*)d_ws;
  float* h0   = ws;                 ws += (long)BATCH * 64 * LSEQ;   // channel-major acts
  float* h1   = ws;                 ws += (long)BATCH * 64 * LSEQ;
  float* pbuf = ws;                 ws += (long)RROWS * 32;
  float* xzb  = ws;                 ws += (long)RROWS * 128;
  float* xcb  = ws;                 ws += (long)RROWS * 64;
  float* xdb  = ws;                 ws += (long)RROWS * 34;
  float* dtb  = ws;                 ws += (long)RROWS * 64;
  float* yb   = ws;                 ws += (long)RROWS * 64;
  float* hfin = ws;                 ws += (long)BATCH * NCH * 1024;
  float* pAb  = ws;                 ws += (long)BATCH * NCH * 1024;
  float* hst  = ws;                 ws += (long)BATCH * NCH * 1024;

  auto gemm = [&](const float* X, long xB, long xL, long xK, int rev,
                  const float* W, const float* bias,
                  float* Y, long yB, long yL, long yO, int K, int O, int act) {
    const dim3 grid(RROWS / 32, (O + 15) / 16), blk(64);
    if (xK == 1) {
      if (K == 32)      gemm_wmma<32, true ><<<grid, blk, 0, stream>>>(X, xB, xL, xK, rev, W, bias, Y, yB, yL, yO, O, act);
      else              gemm_wmma<64, true ><<<grid, blk, 0, stream>>>(X, xB, xL, xK, rev, W, bias, Y, yB, yL, yO, O, act);
    } else {
      if (K == 12)      gemm_wmma<12, false><<<grid, blk, 0, stream>>>(X, xB, xL, xK, rev, W, bias, Y, yB, yL, yO, O, act);
      else              gemm_wmma<64, false><<<grid, blk, 0, stream>>>(X, xB, xL, xK, rev, W, bias, Y, yB, yL, yO, O, act);
    }
  };

  const int eN  = RROWS * DIN;
  const dim3 eGrid((eN + 255) / 256), eBlk(256);
  const dim3 sGrid(BATCH * NCH), sBlk(64);

  const float* hin = x;  // block 0 reads x (B, CIN, L)
  float*       hout = h0;

  for (int blk = 0; blk < NBLK; ++blk) {
    const int  Kp   = (blk == 0) ? C_IN : 2 * HDIM;
    const float* pw = (blk == 0) ? proj1_w : projr_w + (long)(blk - 1) * HDIM * 2 * HDIM;
    const float* pb = (blk == 0) ? proj1_b : projr_b + (long)(blk - 1) * HDIM;

    // p[b,l,o] = W[o,c] * h[b,c,l] + b[o]   (channel-major in, row-major out)
    gemm(hin, (long)Kp * LSEQ, 1, LSEQ, 0, pw, pb,
         pbuf, (long)LSEQ * HDIM, HDIM, 1, Kp, HDIM, 0);

    for (int dir = 0; dir < 2; ++dir) {
      const int m = 2 * blk + dir;
      const float* ipw = in_proj  + (long)m * 128 * 32;
      const float* cw  = conv_w   + (long)m * 64 * 4;
      const float* cb  = conv_b   + (long)m * 64;
      const float* xpw = xproj_w  + (long)m * 34 * 64;
      const float* dtw = dtproj_w + (long)m * 64 * 2;
      const float* dbi = dtproj_b + (long)m * 64;
      const float* Al  = A_log    + (long)m * 1024;
      const float* Dm  = Dparam   + (long)m * 64;
      const float* opw = out_proj + (long)m * 32 * 64;

      // xz = u @ in_proj^T  (u read reversed along L for backward dir)
      gemm(pbuf, (long)LSEQ * HDIM, HDIM, 1, dir, ipw, nullptr,
           xzb, (long)LSEQ * 128, 128, 1, HDIM, 128, 0);

      conv_silu_kernel<<<eGrid, eBlk, 0, stream>>>(xzb, cw, cb, xcb);

      // x_dbl = xc @ xproj^T  (O=34, masked tile)
      gemm(xcb, (long)LSEQ * DIN, DIN, 1, 0, xpw, nullptr,
           xdb, (long)LSEQ * 34, 34, 1, DIN, 34, 0);

      dt_kernel<<<eGrid, eBlk, 0, stream>>>(xdb, dtw, dbi, dtb);

      scan_pass1<<<sGrid, sBlk, 0, stream>>>(dtb, xcb, xdb, Al, hfin, pAb);
      scan_pass2<<<dim3(BATCH * 1024 / 256), dim3(256), 0, stream>>>(hfin, pAb, hst);
      scan_pass3<<<sGrid, sBlk, 0, stream>>>(dtb, xcb, xdb, xzb, Al, Dm, hst, yb);

      // y @ out_proj^T written channel-major into concat slot [dir*32 .. dir*32+31]
      gemm(yb, (long)LSEQ * DIN, DIN, 1, 0, opw, nullptr,
           hout + (long)dir * HDIM * LSEQ, (long)64 * LSEQ, 1, LSEQ, DIN, HDIM, 0);
    }

    hin  = hout;
    hout = (hout == h0) ? h1 : h0;
  }

  // Head: sigmoid(h^T @ Wout^T + bout) -> (B, L, 4)
  gemm(hin, (long)64 * LSEQ, 1, LSEQ, 0, Wout, bout,
       out, (long)LSEQ * C_OUT, C_OUT, 1, 64, C_OUT, 1);
}
